// acLSTM_55611236549289
// MI455X (gfx1250) — matrix operand order repaired and checked by reading, not compiled, not run on hardware
//
#include <hip/hip_runtime.h>

// ---------------------------------------------------------------------------
// Types for CDNA5 WMMA (wave32, v_wmma_f32_16x16x32_bf16)
// ---------------------------------------------------------------------------
typedef __attribute__((ext_vector_type(16))) __bf16 v16bf;
typedef __attribute__((ext_vector_type(8)))  float  v8f;

// ---------------------------------------------------------------------------
// Problem constants
// ---------------------------------------------------------------------------
constexpr int HID   = 1024;
constexpr int FOURH = 4096;
constexpr int BATCH = 128;
constexpr int INR   = 171;   // real input/output feature count
constexpr int KIN   = 192;   // padded to multiple of 32
constexpr int TSTEP = 100;
constexpr int GSTEP = 100;

// ---------------------------------------------------------------------------
// Workspace layout (byte offsets, all 256B aligned)
// Packed weights: per 32(K)x16(N) tile stored row-major (512 bf16), tiles
// ordered [ntile][ktile]. Cell weights are gate-interleaved: packed column
// tile index nt = jtile*4 + gate  (gate order i,f,g,o), j = jtile*16 + col.
// ---------------------------------------------------------------------------
constexpr size_t E_IH1 = (size_t)KIN * FOURH;        // 786432 bf16
constexpr size_t E_HH  = (size_t)HID * FOURH;        // 4194304 bf16
constexpr size_t E_DEC = (size_t)HID * 192;          // 196608 bf16
constexpr size_t E_PW  = E_IH1 + 5 * E_HH + E_DEC;

constexpr size_t OFF_PW   = 0;
constexpr size_t OFF_BIAS = OFF_PW + E_PW * 2;                 // 3*4096 f32
constexpr size_t OFF_BDEC = OFF_BIAS + 3 * 4096 * 4;           // 192 f32
constexpr size_t OFF_XP   = OFF_BDEC + 192 * 4;                // T*B*KIN bf16
constexpr size_t OFF_C    = OFF_XP + (size_t)TSTEP * BATCH * KIN * 2; // 3*B*H f32
constexpr size_t OFF_H0   = OFF_C + 3UL * BATCH * HID * 4;     // 3*B*H bf16
constexpr size_t OFF_H1   = OFF_H0 + 3UL * BATCH * HID * 2;
constexpr size_t OFF_CUR  = OFF_H1 + 3UL * BATCH * HID * 2;    // B*KIN bf16
constexpr size_t ZERO_F32 = (OFF_CUR + (size_t)BATCH * KIN * 2 - OFF_C) / 4;

// ---------------------------------------------------------------------------
// Helpers
// ---------------------------------------------------------------------------
__device__ __forceinline__ unsigned short f2bf(float f) {
    union { float f; unsigned u; } a; a.f = f;
    unsigned u = a.u;
    unsigned r = u + 0x7FFFu + ((u >> 16) & 1u);   // round-to-nearest-even
    return (unsigned short)(r >> 16);
}

__device__ __forceinline__ float sigmoidf(float x) {
    return 1.0f / (1.0f + __expf(-x));
}

union Frag { v16bf v; uint4 q[2]; };

// A-fragment: row-major bf16 activations, ld = K stride (elements).
// lanes 0-15: K = {k0..k0+7, k0+16..k0+23}; lanes 16-31: +8.
__device__ __forceinline__ v16bf load_a(const unsigned short* base, int ld,
                                        int row, int k0, int lane) {
    Frag f;
    const unsigned short* p = base + (size_t)row * ld + k0 + ((lane >> 4) << 3);
    f.q[0] = *(const uint4*)(p);
    f.q[1] = *(const uint4*)(p + 16);
    return f.v;
}

// B-fragment: packed tile row-major [32][16]; lane L holds column N=L%16,
// K = {8h..8h+7, 16+8h..16+8h+7} with h = L/16 (same order as A).
__device__ __forceinline__ v16bf load_b(const unsigned short* tile, int lane) {
    union { v16bf v; unsigned short u[16]; } f;
    const int n = lane & 15, h = lane >> 4;
#pragma unroll
    for (int s = 0; s < 16; ++s) {
        const int k = (s < 8) ? (8 * h + s) : (16 + 8 * h + (s - 8));
        f.u[s] = tile[k * 16 + n];
    }
    return f.v;
}

#define WMMA_BF16(a, b, c) \
    __builtin_amdgcn_wmma_f32_16x16x32_bf16(false, (a), false, (b), (short)0, (c), false, false)

// ---------------------------------------------------------------------------
// Packing kernels (run once per launch; all elements written -> deterministic)
// ---------------------------------------------------------------------------
__global__ void pack_w_gates_kernel(const float* __restrict__ W,
                                    unsigned short* __restrict__ out,
                                    int Ktiles, int Kreal) {
    size_t idx = (size_t)blockIdx.x * blockDim.x + threadIdx.x;
    size_t total = (size_t)Ktiles * 32 * FOURH;
    if (idx >= total) return;
    int col = (int)(idx & 15);
    int k   = (int)((idx >> 4) & 31);
    size_t t = idx >> 9;
    int kt = (int)(t % Ktiles);
    int nt = (int)(t / Ktiles);
    int gate = nt & 3, jtile = nt >> 2;
    int n  = gate * HID + jtile * 16 + col;   // original row of W (4H, K)
    int kk = kt * 32 + k;
    float v = (kk < Kreal) ? W[(size_t)n * Kreal + kk] : 0.0f;
    out[idx] = f2bf(v);
}

__global__ void pack_dec_kernel(const float* __restrict__ W,
                                unsigned short* __restrict__ out) {
    size_t idx = (size_t)blockIdx.x * blockDim.x + threadIdx.x;
    if (idx >= E_DEC) return;
    int col = (int)(idx & 15);
    int k   = (int)((idx >> 4) & 31);
    size_t t = idx >> 9;
    int kt = (int)(t % 32);
    int nt = (int)(t / 32);
    int n  = nt * 16 + col;
    int kk = kt * 32 + k;
    float v = (n < INR) ? W[(size_t)n * HID + kk] : 0.0f;
    out[idx] = f2bf(v);
}

__global__ void pack_bias_kernel(const float* bi1, const float* bh1,
                                 const float* bi2, const float* bh2,
                                 const float* bi3, const float* bh3,
                                 const float* bdec_in,
                                 float* __restrict__ bias_out,
                                 float* __restrict__ bdec_out) {
    int idx = blockIdx.x * blockDim.x + threadIdx.x;
    if (idx < 3 * FOURH) {
        int l = idx / FOURH, np = idx % FOURH;
        int nt = np >> 4, col = np & 15;
        int n = (nt & 3) * HID + (nt >> 2) * 16 + col;
        const float* a = (l == 0) ? bi1 : ((l == 1) ? bi2 : bi3);
        const float* b = (l == 0) ? bh1 : ((l == 1) ? bh2 : bh3);
        bias_out[idx] = a[n] + b[n];
    } else if (idx < 3 * FOURH + 192) {
        int n = idx - 3 * FOURH;
        bdec_out[n] = (n < INR) ? bdec_in[n] : 0.0f;
    }
}

__global__ void pack_x_kernel(const float* __restrict__ seq,
                              unsigned short* __restrict__ xp) {
    size_t idx = (size_t)blockIdx.x * blockDim.x + threadIdx.x;
    size_t total = (size_t)TSTEP * BATCH * KIN;
    if (idx >= total) return;
    int o = (int)(idx % KIN);
    size_t r = idx / KIN;
    int b = (int)(r % BATCH);
    int t = (int)(r / BATCH);
    float v = (o < INR) ? seq[((size_t)b * TSTEP + t) * INR + o] : 0.0f;
    xp[idx] = f2bf(v);
}

__global__ void zero_kernel(float* __restrict__ p, size_t n) {
    size_t idx = (size_t)blockIdx.x * blockDim.x + threadIdx.x;
    if (idx < n) p[idx] = 0.0f;
}

// ---------------------------------------------------------------------------
// One full LSTM timestep: all 3 cells (they only read OLD state -> parallel).
// grid = (64 jtiles, 3 layers), block = 256 (8 waves; wave w -> batch rows
// 16w..16w+15). Each wave's 4 accumulators = gates (i,f,g,o) of 16 hidden
// units -> pointwise LSTM done entirely in registers.
// ---------------------------------------------------------------------------
__global__ __launch_bounds__(256) void lstm_step_kernel(
    const char* __restrict__ ws,
    const unsigned short* __restrict__ x_in,     // [B, KIN] bf16 (layer-0 input)
    const unsigned short* __restrict__ h_old,    // [3][B][H] bf16
    unsigned short* __restrict__ h_new)          // [3][B][H] bf16
{
    const int l    = blockIdx.y;
    const int jt   = blockIdx.x;
    const int lane = threadIdx.x & 31;
    const int wv   = threadIdx.x >> 5;
    const int m0   = wv * 16;
    const int row  = m0 + (lane & 15);

    const unsigned short* pw = (const unsigned short*)(ws + OFF_PW);
    const unsigned short* wih =
        (l == 0) ? pw : pw + E_IH1 + (size_t)(2 * l - 1) * E_HH;
    const unsigned short* whh = pw + E_IH1 + (size_t)(2 * l) * E_HH;

    v8f zf = {};
    v8f acc[4] = { zf, zf, zf, zf };

    // ---- input contribution:  x @ W_ih^T  (layer0)  or  h_old[l-1] @ W_ih^T
    {
        const unsigned short* ain;
        int Kt;
        if (l == 0) { ain = x_in; Kt = KIN / 32; }
        else        { ain = h_old + (size_t)(l - 1) * BATCH * HID; Kt = HID / 32; }
        const int ld = Kt * 32;
        for (int kt = 0; kt < Kt; ++kt) {
            v16bf a = load_a(ain, ld, row, kt * 32, lane);
#pragma unroll
            for (int g = 0; g < 4; ++g) {
                const unsigned short* tile =
                    wih + ((size_t)(jt * 4 + g) * Kt + kt) * 512;
                acc[g] = WMMA_BF16(a, load_b(tile, lane), acc[g]);
            }
        }
    }

    // ---- recurrent contribution: h_old[l] @ W_hh^T
    {
        const unsigned short* ain = h_old + (size_t)l * BATCH * HID;
        for (int kt = 0; kt < HID / 32; ++kt) {
            v16bf a = load_a(ain, HID, row, kt * 32, lane);
#pragma unroll
            for (int g = 0; g < 4; ++g) {
                const unsigned short* tile =
                    whh + ((size_t)(jt * 4 + g) * (HID / 32) + kt) * 512;
                acc[g] = WMMA_BF16(a, load_b(tile, lane), acc[g]);
            }
        }
    }

    // ---- bias + pointwise LSTM (C/D layout: vgpr r, lanes0-15: M=r, 16-31: M=r+8)
    const float* bias = (const float*)(ws + OFF_BIAS) + (size_t)l * FOURH;
    const int col = lane & 15;
    const int j   = jt * 16 + col;
    const float bi = bias[(jt * 4 + 0) * 16 + col];
    const float bf = bias[(jt * 4 + 1) * 16 + col];
    const float bg = bias[(jt * 4 + 2) * 16 + col];
    const float bo = bias[(jt * 4 + 3) * 16 + col];

    float* cst = (float*)(ws + OFF_C) + (size_t)l * BATCH * HID;
    unsigned short* hn = h_new + (size_t)l * BATCH * HID;
    const int rbase = m0 + ((lane >> 4) << 3);

#pragma unroll
    for (int r = 0; r < 8; ++r) {
        const int m = rbase + r;
        float gi = sigmoidf(acc[0][r] + bi);
        float gf = sigmoidf(acc[1][r] + bf);
        float gg = tanhf(acc[2][r] + bg);
        float go = sigmoidf(acc[3][r] + bo);
        size_t idx = (size_t)m * HID + j;
        float cn = gf * cst[idx] + gi * gg;
        cst[idx] = cn;
        hn[idx]  = f2bf(go * tanhf(cn));
    }
}

// ---------------------------------------------------------------------------
// Decoder: out = h2 @ W_dec^T + b_dec ; writes d_out[b][gstep][o] (gstep>=0)
// and the bf16 `cur` buffer (next autoregressive input).
// grid = 12 ntiles, block = 256 (8 waves over the 128 batch rows).
// ---------------------------------------------------------------------------
__global__ __launch_bounds__(256) void decoder_kernel(
    const char* __restrict__ ws,
    const unsigned short* __restrict__ h2,       // [B][H] bf16
    float* __restrict__ out, int gstep,
    unsigned short* __restrict__ cur)            // [B][KIN] bf16
{
    const int nt   = blockIdx.x;
    const int lane = threadIdx.x & 31;
    const int wv   = threadIdx.x >> 5;
    const int m0   = wv * 16;
    const int row  = m0 + (lane & 15);

    const unsigned short* wdec =
        (const unsigned short*)(ws + OFF_PW) + E_IH1 + 5 * E_HH;

    v8f acc = {};
    for (int kt = 0; kt < HID / 32; ++kt) {
        v16bf a = load_a(h2, HID, row, kt * 32, lane);
        const unsigned short* tile = wdec + ((size_t)nt * (HID / 32) + kt) * 512;
        acc = WMMA_BF16(a, load_b(tile, lane), acc);
    }

    const float* bd = (const float*)(ws + OFF_BDEC);
    const int col = nt * 16 + (lane & 15);
    const float bv = bd[col];
    const int rbase = m0 + ((lane >> 4) << 3);

#pragma unroll
    for (int r = 0; r < 8; ++r) {
        const int m = rbase + r;
        float v = acc[r] + bv;
        if (col < INR) {
            cur[(size_t)m * KIN + col] = f2bf(v);
            if (gstep >= 0)
                out[(size_t)m * GSTEP * INR + (size_t)gstep * INR + col] = v;
        }
    }
}

// ---------------------------------------------------------------------------
// Host-side orchestration (graph-capture friendly: only kernel launches)
// ---------------------------------------------------------------------------
extern "C" void kernel_launch(void* const* d_in, const int* in_sizes, int n_in,
                              void* d_out, int out_size, void* d_ws, size_t ws_size,
                              hipStream_t stream) {
    (void)in_sizes; (void)n_in; (void)out_size; (void)ws_size;
    char* ws = (char*)d_ws;

    const float* seq  = (const float*)d_in[0];
    const float* Wih1 = (const float*)d_in[1];
    const float* Whh1 = (const float*)d_in[2];
    const float* bih1 = (const float*)d_in[3];
    const float* bhh1 = (const float*)d_in[4];
    const float* Wih2 = (const float*)d_in[5];
    const float* Whh2 = (const float*)d_in[6];
    const float* bih2 = (const float*)d_in[7];
    const float* bhh2 = (const float*)d_in[8];
    const float* Wih3 = (const float*)d_in[9];
    const float* Whh3 = (const float*)d_in[10];
    const float* bih3 = (const float*)d_in[11];
    const float* bhh3 = (const float*)d_in[12];
    const float* Wdec = (const float*)d_in[13];
    const float* bdec = (const float*)d_in[14];

    unsigned short* pw  = (unsigned short*)(ws + OFF_PW);
    unsigned short* xp  = (unsigned short*)(ws + OFF_XP);
    unsigned short* cur = (unsigned short*)(ws + OFF_CUR);
    unsigned short* Hb[2] = { (unsigned short*)(ws + OFF_H0),
                              (unsigned short*)(ws + OFF_H1) };

    // ---- pack weights / bias / inputs into WMMA-friendly bf16 layout
    {
        size_t tot = E_IH1;
        pack_w_gates_kernel<<<(tot + 255) / 256, 256, 0, stream>>>(Wih1, pw, KIN / 32, INR);
    }
    const float* WHs[5] = { Whh1, Wih2, Whh2, Wih3, Whh3 };
    for (int i = 0; i < 5; ++i) {
        unsigned short* dst = pw + E_IH1 + (size_t)i * E_HH;
        pack_w_gates_kernel<<<(E_HH + 255) / 256, 256, 0, stream>>>(WHs[i], dst, HID / 32, HID);
    }
    pack_dec_kernel<<<(E_DEC + 255) / 256, 256, 0, stream>>>(Wdec, pw + E_IH1 + 5 * E_HH);
    pack_bias_kernel<<<(3 * FOURH + 192 + 255) / 256, 256, 0, stream>>>(
        bih1, bhh1, bih2, bhh2, bih3, bhh3, bdec,
        (float*)(ws + OFF_BIAS), (float*)(ws + OFF_BDEC));
    {
        size_t tot = (size_t)TSTEP * BATCH * KIN;
        pack_x_kernel<<<(tot + 255) / 256, 256, 0, stream>>>(seq, xp);
    }
    zero_kernel<<<(ZERO_F32 + 255) / 256, 256, 0, stream>>>((float*)(ws + OFF_C), ZERO_F32);

    // ---- warmup (conditioning) pass over the seed sequence
    for (int t = 0; t < TSTEP; ++t) {
        lstm_step_kernel<<<dim3(64, 3), 256, 0, stream>>>(
            ws, xp + (size_t)t * BATCH * KIN, Hb[t & 1], Hb[1 - (t & 1)]);
    }
    // produce first autoregressive input (final warmup `out`); not recorded
    decoder_kernel<<<12, 256, 0, stream>>>(
        ws, Hb[0] + 2UL * BATCH * HID, (float*)d_out, -1, cur);

    // ---- autoregressive generation
    for (int g = 0; g < GSTEP; ++g) {
        int t = TSTEP + g;
        lstm_step_kernel<<<dim3(64, 3), 256, 0, stream>>>(
            ws, cur, Hb[t & 1], Hb[1 - (t & 1)]);
        decoder_kernel<<<12, 256, 0, stream>>>(
            ws, Hb[1 - (t & 1)] + 2UL * BATCH * HID, (float*)d_out, g, cur);
    }
}
